// DualStreamBlock_3908420240260
// MI455X (gfx1250) — 16-bit and 8-bit matrix operand order repaired and checked by reading, compiled, not run on hardware
//
#include <hip/hip_runtime.h>
#include <stdint.h>

typedef __attribute__((ext_vector_type(8)))  int      v8i;
typedef __attribute__((ext_vector_type(8)))  float    v8f;
typedef __attribute__((ext_vector_type(16))) _Float16 v16h;

#define DEV __device__ __forceinline__

DEV float sigmoidf_(float x) { return 1.f / (1.f + __expf(-x)); }
DEV float siluf_(float x)    { return x * sigmoidf_(x); }

DEV float block_reduce_sum(float v, float* sh) {
  const int tid = threadIdx.x;
#pragma unroll
  for (int o = 16; o > 0; o >>= 1) v += __shfl_down(v, o, 32);
  __syncthreads();
  if ((tid & 31) == 0) sh[tid >> 5] = v;
  __syncthreads();
  float r = sh[0];
#pragma unroll
  for (int i = 1; i < 8; ++i) r += sh[i];
  return r;
}
DEV float block_reduce_max(float v, float* sh) {
  const int tid = threadIdx.x;
#pragma unroll
  for (int o = 16; o > 0; o >>= 1) v = fmaxf(v, __shfl_down(v, o, 32));
  __syncthreads();
  if ((tid & 31) == 0) sh[tid >> 5] = v;
  __syncthreads();
  float r = sh[0];
#pragma unroll
  for (int i = 1; i < 8; ++i) r = fmaxf(r, sh[i]);
  return r;
}

__global__ void absmean_partial_kernel(const float* __restrict__ W, size_t n,
                                       float* __restrict__ accum) {
  __shared__ float sh[8];
  float s = 0.f;
  for (size_t i = (size_t)blockIdx.x * 256 + threadIdx.x; i < n;
       i += (size_t)gridDim.x * 256)
    s += fabsf(W[i]);
  s = block_reduce_sum(s, sh);
  if (threadIdx.x == 0) atomicAdd(accum, s);
}

__global__ void ternarize_pack_kernel(const float* __restrict__ W, int Kd,
                                      int Od, const float* __restrict__ accum,
                                      float* __restrict__ wsc,
                                      int8_t* __restrict__ Wpk) {
  const size_t n = (size_t)Kd * Od;
  const float m = fmaxf(accum[0] / (float)n, 1e-5f);
  const int kcnt = Kd >> 6;
  size_t i = (size_t)blockIdx.x * 256 + threadIdx.x;
  if (i == 0) wsc[0] = m;
  for (; i < n; i += (size_t)gridDim.x * 256) {
    float q = rintf(W[i] / m);
    q = fminf(1.f, fmaxf(-1.f, q));
    const int k = (int)(i / Od), o = (int)(i % Od);
    const int co = o >> 4, lmn = o & 15;
    const int kc = k >> 6, kin = k & 63;
    const int byt = kin & 3;
    const int lh = (kin >> 3) & 1;
    const int j = 2 * (kin >> 4) + ((kin >> 2) & 1);
    const size_t off =
        (((size_t)co * kcnt + kc) * 32 + (lmn + 16 * lh)) * 32 + j * 4 + byt;
    Wpk[off] = (int8_t)q;
  }
}

__global__ void rms_quant_kernel(const float* __restrict__ X, int K,
                                 int8_t* __restrict__ Xq, float* __restrict__ asc,
                                 float eps) {
  __shared__ float sh[8];
  const int t = blockIdx.x, tid = threadIdx.x;
  const float* x = X + (size_t)t * K;
  float ss = 0.f;
  for (int k = tid; k < K; k += 256) { float v = x[k]; ss += v * v; }
  ss = block_reduce_sum(ss, sh);
  const float rs = rsqrtf(ss / (float)K + eps);
  float amax = 0.f;
  for (int k = tid; k < K; k += 256) amax = fmaxf(amax, fabsf(x[k] * rs));
  amax = block_reduce_max(amax, sh);
  amax = fmaxf(amax, 1e-5f);
  const float s = 127.f / amax;
  for (int k = tid; k < K; k += 256) {
    float q = rintf(x[k] * rs * s);
    q = fminf(127.f, fmaxf(-128.f, q));
    Xq[(size_t)t * K + k] = (int8_t)q;
  }
  if (tid == 0) asc[t] = amax / 127.f;
}

template <int MODE>
__global__ void bitlin_gemm_iu8(const int8_t* __restrict__ Aq,
                                const float* __restrict__ ascale,
                                const int8_t* __restrict__ Wpk,
                                const float* __restrict__ wscale_p,
                                float* __restrict__ Out, int T, int K, int O) {
  const int lane = threadIdx.x & 31, wave = threadIdx.x >> 5;
  const int lm = lane & 15, lh = lane >> 4;
  const int row0 = blockIdx.x * 32 + (wave >> 2) * 16;
  const int col0 = blockIdx.y * 256 + (wave & 3) * 64;
  if (row0 >= T || col0 >= O) return;
  const int kcnt = K >> 6;
  const int8_t* arow = Aq + (size_t)(row0 + lm) * K;
  const int8_t* bbase[4];
#pragma unroll
  for (int c = 0; c < 4; ++c)
    bbase[c] = Wpk + (((size_t)((col0 >> 4) + c) * kcnt) * 32 + lane) * 32;
  v8i acc[4] = {};
#pragma unroll 2
  for (int kc = 0; kc < kcnt; ++kc) {
    const int kb = kc * 64 + lh * 8;
    v8i a;
#pragma unroll
    for (int j = 0; j < 8; ++j)
      a[j] = *(const int*)(arow + kb + (j >> 1) * 16 + (j & 1) * 4);
    __builtin_prefetch(arow + kb + 64, 0, 1);
#pragma unroll
    for (int c = 0; c < 4; ++c) {
      const v8i bf = *(const v8i*)(bbase[c] + (size_t)kc * 1024);
      __builtin_prefetch(bbase[c] + (size_t)(kc + 1) * 1024, 0, 1);
      acc[c] = __builtin_amdgcn_wmma_i32_16x16x64_iu8(true, a, true, bf,
                                                      acc[c], false, false);
    }
  }
  const float ws = wscale_p[0];
  float as[8];
#pragma unroll
  for (int r = 0; r < 8; ++r) as[r] = ascale[row0 + r + 8 * lh] * ws;
#pragma unroll
  for (int c = 0; c < 4; ++c) {
    const int col = col0 + c * 16 + lm;
#pragma unroll
    for (int r = 0; r < 8; ++r) {
      const size_t idx = (size_t)(row0 + r + 8 * lh) * O + col;
      const float v = (float)acc[c][r] * as[r];
      if (MODE == 1) { const float g = Out[idx]; Out[idx] = siluf_(g) * v; }
      else Out[idx] = v;
    }
  }
}

__global__ void ln_modulate_kernel(const float* __restrict__ X,
                                   const float* __restrict__ mods, int shift_c,
                                   int scale_c, float* __restrict__ Y, int S) {
  __shared__ float sh[8];
  const int t = blockIdx.x, tid = threadIdx.x;
  const int b = t / S;
  const float* x = X + (size_t)t * 768;
  float s1 = 0.f, s2 = 0.f;
  for (int k = tid; k < 768; k += 256) { float v = x[k]; s1 += v; s2 += v * v; }
  s1 = block_reduce_sum(s1, sh);
  s2 = block_reduce_sum(s2, sh);
  const float mu = s1 / 768.f;
  const float rs = rsqrtf(s2 / 768.f - mu * mu + 1e-6f);
  const float* shp = mods + (size_t)b * 11520 + shift_c * 768;
  const float* scp = mods + (size_t)b * 11520 + scale_c * 768;
  for (int k = tid; k < 768; k += 256)
    Y[(size_t)t * 768 + k] = (x[k] - mu) * rs * (1.f + scp[k]) + shp[k];
}

__global__ void rms_gate_kernel(const float* __restrict__ Hb,
                                float* __restrict__ G,
                                const float* __restrict__ gnorm) {
  __shared__ float sh[8];
  const int t = blockIdx.x, tid = threadIdx.x;
  const float* h = Hb + (size_t)t * 768;
  float ss = 0.f;
  for (int k = tid; k < 768; k += 256) { float v = h[k]; ss += v * v; }
  ss = block_reduce_sum(ss, sh);
  const float rs = rsqrtf(ss / 768.f + 1e-6f);
  for (int k = tid; k < 768; k += 256) {
    const size_t i = (size_t)t * 768 + k;
    G[i] = h[k] * rs * gnorm[k] * siluf_(G[i]);
  }
}

__global__ void rmsnormw_kernel(const float* __restrict__ X,
                                const float* __restrict__ w,
                                float* __restrict__ Y, int Kd) {
  __shared__ float sh[8];
  const int b = blockIdx.x, tid = threadIdx.x;
  const float* x = X + (size_t)b * Kd;
  float ss = 0.f;
  for (int k = tid; k < Kd; k += 256) { float v = x[k]; ss += v * v; }
  ss = block_reduce_sum(ss, sh);
  const float rs = rsqrtf(ss / (float)Kd + 1e-6f);
  for (int k = tid; k < Kd; k += 256) Y[(size_t)b * Kd + k] = x[k] * rs * w[k];
}

__global__ void matvec_kernel(const float* __restrict__ in,
                              const float* __restrict__ W,
                              const float* __restrict__ bias,
                              float* __restrict__ out, int Kd, int Od) {
  const int o = blockIdx.x * 256 + threadIdx.x;
  const int b = blockIdx.y;
  if (o >= Od) return;
  const float* xr = in + (size_t)b * Kd;
  float s = bias ? bias[o] : 0.f;
  for (int k = 0; k < Kd; ++k) s += xr[k] * W[(size_t)k * Od + o];
  out[(size_t)b * Od + o] = s;
}

__global__ void silu_kernel(const float* X, float* Y, size_t n) {
  size_t i = (size_t)blockIdx.x * 256 + threadIdx.x;
  if (i < n) Y[i] = siluf_(X[i]);
}
__global__ void sigmoid_inplace_kernel(float* X, size_t n) {
  size_t i = (size_t)blockIdx.x * 256 + threadIdx.x;
  if (i < n) X[i] = sigmoidf_(X[i]);
}
__global__ void gate_combine_kernel(float* I, const float* F, size_t n) {
  size_t i = (size_t)blockIdx.x * 256 + threadIdx.x;
  if (i < n) I[i] = siluf_(I[i]) * (1.f - F[i]);
}
__global__ void glu_ada_kernel(const float* Y2, float* Hg, size_t n) {
  size_t i = (size_t)blockIdx.x * 256 + threadIdx.x;
  if (i < n) {
    size_t b = i / 2048, j = i % 2048;
    Hg[i] = siluf_(Y2[b * 4096 + j]) * Y2[b * 4096 + 2048 + j];
  }
}
__global__ void residual_gate_kernel(float* X, const float* A,
                                     const float* mods, int gc, int S,
                                     size_t n) {
  size_t i = (size_t)blockIdx.x * 256 + threadIdx.x;
  if (i < n) {
    size_t t = i / 768;
    int d = (int)(i % 768);
    int b = (int)(t / S);
    X[i] += mods[(size_t)b * 11520 + gc * 768 + d] * A[i];
  }
}

__global__ void hgrn_chunk_scan_kernel(const float* __restrict__ F,
                                       float* __restrict__ IH,
                                       float* __restrict__ Pprod, int Bn, int T,
                                       int CH) {
  const int idx = blockIdx.x * 256 + threadIdx.x;
  const int total = Bn * 768 * CH;
  if (idx >= total) return;
  const int c = idx % CH, bd = idx / CH;
  const int d = bd % 768, b = bd / 768;
  const int L = T / CH;
  const size_t base = ((size_t)b * T + (size_t)c * L) * 768 + d;
  const float* f = F + base;
  float* ih = IH + base;
  float h = 0.f, p = 1.f;
  for (int t = 0; t < L; ++t) {
    const size_t o = (size_t)t * 768;
    const float ft = f[o];
    p *= ft;
    h = ft * h + ih[o];
    ih[o] = h;
  }
  Pprod[idx] = p;
}
__global__ void hgrn_carry_kernel(const float* __restrict__ IH,
                                  const float* __restrict__ Pprod,
                                  float* __restrict__ Carry, int Bn, int T,
                                  int CH) {
  const int bd = blockIdx.x * 256 + threadIdx.x;
  if (bd >= Bn * 768) return;
  const int d = bd % 768, b = bd / 768;
  const int L = T / CH;
  float h = 0.f;
  for (int c = 0; c < CH; ++c) {
    Carry[(size_t)bd * CH + c] = h;
    const float last = IH[((size_t)b * T + (size_t)c * L + (L - 1)) * 768 + d];
    h = h * Pprod[(size_t)bd * CH + c] + last;
  }
}
__global__ void hgrn_apply_kernel(const float* __restrict__ F,
                                  float* __restrict__ IH,
                                  const float* __restrict__ Carry, int Bn,
                                  int T, int CH) {
  const int idx = blockIdx.x * 256 + threadIdx.x;
  const int total = Bn * 768 * CH;
  if (idx >= total) return;
  const int c = idx % CH, bd = idx / CH;
  const int d = bd % 768, b = bd / 768;
  const int L = T / CH;
  const float cr = Carry[(size_t)bd * CH + c];
  if (cr == 0.f) return;
  const size_t base = ((size_t)b * T + (size_t)c * L) * 768 + d;
  const float* f = F + base;
  float* ih = IH + base;
  float p = 1.f;
  for (int t = 0; t < L; ++t) {
    const size_t o = (size_t)t * 768;
    p *= f[o];
    ih[o] += cr * p;
  }
}

__global__ void to_f16_heads_kernel(const float* __restrict__ X,
                                    _Float16* __restrict__ Y, int S, size_t n) {
  size_t i = (size_t)blockIdx.x * 256 + threadIdx.x;
  if (i >= n) return;
  size_t t = i / 768;
  int c = (int)(i % 768);
  int b = (int)(t / S), s = (int)(t % S);
  int h = c / 64, d = c % 64;
  Y[((size_t)(b * 12 + h) * S + s) * 64 + d] = (_Float16)X[i];
}
__global__ void to_f16_headsT_kernel(const float* __restrict__ X,
                                     _Float16* __restrict__ Y, int S,
                                     size_t n) {
  size_t i = (size_t)blockIdx.x * 256 + threadIdx.x;
  if (i >= n) return;
  size_t t = i / 768;
  int c = (int)(i % 768);
  int b = (int)(t / S), s = (int)(t % S);
  int h = c / 64, d = c % 64;
  Y[((size_t)(b * 12 + h) * 64 + d) * S + s] = (_Float16)X[i];
}

__global__ void cross_attn_kernel(const _Float16* __restrict__ Qh,
                                  const _Float16* __restrict__ Kh,
                                  const _Float16* __restrict__ Vt,
                                  float* __restrict__ Ob, int Bn, int Nq,
                                  int Mk) {
  __shared__ float    Sbuf[8][16][33];
  __shared__ _Float16 Pbuf[8][16][36];
  __shared__ float    mrow[8][16], lrow[8][16], arw[8][16];
  const int lane = threadIdx.x & 31, wid = threadIdx.x >> 5;
  const int lm = lane & 15, lh = lane >> 4;
  const int ntiles = Nq >> 4;
  const int gw = blockIdx.x * 8 + wid;
  const int nt = gw % ntiles;
  const int h = (gw / ntiles) % 12;
  const int b = gw / (ntiles * 12);
  if (b >= Bn) return;
  const _Float16* qrow = Qh + ((size_t)(b * 12 + h) * Nq + nt * 16 + lm) * 64;
  v16h qa0, qa1;
#pragma unroll
  for (int e = 0; e < 16; ++e) {
    const int K0 = lh * 16 + e;
    qa0[e] = qrow[K0];
    qa1[e] = qrow[32 + K0];
  }
  v8f oacc[4] = {};
  if (lane < 16) { mrow[wid][lane] = -1e30f; lrow[wid][lane] = 0.f; }
  for (int mt = 0; mt < (Mk >> 5); ++mt) {
#pragma unroll
    for (int s = 0; s < 2; ++s) {
      const _Float16* krow =
          Kh + ((size_t)(b * 12 + h) * Mk + mt * 32 + s * 16 + lm) * 64;
      v16h kb0, kb1;
#pragma unroll
      for (int e = 0; e < 16; ++e) {
        kb0[e] = krow[lh * 16 + e];
        kb1[e] = krow[32 + lh * 16 + e];
      }
      v8f sacc = {};
      sacc = __builtin_amdgcn_wmma_f32_16x16x32_f16(false, qa0, false, kb0,
                                                    (short)0, sacc, false, false);
      sacc = __builtin_amdgcn_wmma_f32_16x16x32_f16(false, qa1, false, kb1,
                                                    (short)0, sacc, false, false);
#pragma unroll
      for (int r = 0; r < 8; ++r)
        Sbuf[wid][r + 8 * lh][s * 16 + lm] = sacc[r] * 0.125f;
    }
    if (lane < 16) {
      float mx = mrow[wid][lane];
#pragma unroll 4
      for (int j = 0; j < 32; ++j) mx = fmaxf(mx, Sbuf[wid][lane][j]);
      const float al = __expf(mrow[wid][lane] - mx);
      float sum = 0.f;
#pragma unroll 4
      for (int j = 0; j < 32; ++j) {
        const float p = __expf(Sbuf[wid][lane][j] - mx);
        Pbuf[wid][lane][j] = (_Float16)p;
        sum += p;
      }
      mrow[wid][lane] = mx;
      lrow[wid][lane] = lrow[wid][lane] * al + sum;
      arw[wid][lane] = al;
    }
#pragma unroll
    for (int c = 0; c < 4; ++c)
#pragma unroll
      for (int r = 0; r < 8; ++r) oacc[c][r] *= arw[wid][r + 8 * lh];
    v16h pa;
#pragma unroll
    for (int e = 0; e < 16; ++e)
      pa[e] = Pbuf[wid][lm][lh * 16 + e];
#pragma unroll
    for (int c = 0; c < 4; ++c) {
      const _Float16* vrow =
          Vt + ((size_t)(b * 12 + h) * 64 + c * 16 + lm) * Mk + mt * 32 +
          lh * 16;
      v16h vb;
#pragma unroll
      for (int e = 0; e < 16; ++e) vb[e] = vrow[e];
      oacc[c] = __builtin_amdgcn_wmma_f32_16x16x32_f16(false, pa, false, vb,
                                                       (short)0, oacc[c], false,
                                                       false);
    }
  }
#pragma unroll
  for (int c = 0; c < 4; ++c)
#pragma unroll
    for (int r = 0; r < 8; ++r) {
      const int n = nt * 16 + r + 8 * lh;
      Ob[((size_t)(b * Nq + n)) * 768 + h * 64 + c * 16 + lm] =
          oacc[c][r] / lrow[wid][r + 8 * lh];
    }
}

extern "C" void kernel_launch(void* const* d_in, const int* in_sizes, int n_in,
                              void* d_out, int out_size, void* d_ws,
                              size_t ws_size, hipStream_t stream) {
  constexpr int Bb = 4, Nn = 2048, Mm = 512;
  constexpr int TX = Bb * Nn;
  constexpr int TY = Bb * Mm;
  constexpr int CH = 16;
  (void)in_sizes; (void)n_in; (void)out_size; (void)ws_size;

  const float* x_in = (const float*)d_in[0];
  const float* y_in = (const float*)d_in[1];
  const float* c_in = (const float*)d_in[2];
  const float* ada_w_in  = (const float*)d_in[3];
  const float* ada_wgate = (const float*)d_in[4];
  const float* ada_wdown = (const float*)d_in[5];
  const float* ada_woutp = (const float*)d_in[6];
  const float* ada_boutp = (const float*)d_in[7];
  const float* ada_normw = (const float*)d_in[8];
  const float* ada_wout2 = (const float*)d_in[9];
  const float* ada_bout2 = (const float*)d_in[10];
  const float* gnorm_x = (const float*)d_in[15];
  const float* gnorm_y = (const float*)d_in[20];

  struct QW { const float* w; int K, O; int8_t* wt; };
  QW qw[16] = {
      {(const float*)d_in[11], 768, 768, nullptr},
      {(const float*)d_in[12], 768, 768, nullptr},
      {(const float*)d_in[13], 768, 768, nullptr},
      {(const float*)d_in[14], 768, 768, nullptr},
      {(const float*)d_in[16], 768, 768, nullptr},
      {(const float*)d_in[17], 768, 768, nullptr},
      {(const float*)d_in[18], 768, 768, nullptr},
      {(const float*)d_in[19], 768, 768, nullptr},
      {(const float*)d_in[21], 768, 768, nullptr},
      {(const float*)d_in[22], 768, 768, nullptr},
      {(const float*)d_in[23], 768, 768, nullptr},
      {(const float*)d_in[24], 768, 768, nullptr},
      {(const float*)d_in[25], 768, 6144, nullptr},
      {(const float*)d_in[26], 3072, 768, nullptr},
      {(const float*)d_in[27], 768, 6144, nullptr},
      {(const float*)d_in[28], 3072, 768, nullptr},
  };

  char* ws = (char*)d_ws;
  size_t off = 0;
  auto alloc = [&](size_t bytes) -> void* {
    void* p = ws + off;
    off += (bytes + 255) & ~(size_t)255;
    return p;
  };

  float* accums  = (float*)alloc(16 * sizeof(float));
  float* wscales = (float*)alloc(16 * sizeof(float));
  for (int i = 0; i < 16; ++i)
    qw[i].wt = (int8_t*)alloc((size_t)qw[i].K * qw[i].O);

  float* cs  = (float*)alloc((size_t)Bb * 768 * 4);
  float* h1  = (float*)alloc((size_t)Bb * 768 * 4);
  float* y2  = (float*)alloc((size_t)Bb * 4096 * 4);
  float* h2g = (float*)alloc((size_t)Bb * 2048 * 4);
  float* h2  = (float*)alloc((size_t)Bb * 768 * 4);
  float* h3  = (float*)alloc((size_t)Bb * 11520 * 4);
  float* h4  = (float*)alloc((size_t)Bb * 11520 * 4);
  float* mods = (float*)alloc((size_t)Bb * 11520 * 4);

  float* bufXM = (float*)alloc((size_t)TX * 768 * 4);
  float* bufI  = (float*)alloc((size_t)TX * 768 * 4);
  float* bufF  = (float*)alloc((size_t)TX * 768 * 4);
  float* bufG  = (float*)alloc((size_t)TX * 768 * 4);
  float* bufBig = (float*)alloc((size_t)TX * 3072 * 4);
  float* bufK  = (float*)alloc((size_t)TY * 768 * 4);
  float* bufV  = (float*)alloc((size_t)TY * 768 * 4);
  int8_t* xq = (int8_t*)alloc((size_t)TX * 3072);
  float* asc = (float*)alloc((size_t)TX * 4);
  _Float16* qh = (_Float16*)alloc((size_t)TX * 768 * 2);
  _Float16* kh = (_Float16*)alloc((size_t)TY * 768 * 2);
  _Float16* vh = (_Float16*)alloc((size_t)TY * 768 * 2);
  float* scanP = (float*)alloc((size_t)Bb * 768 * CH * 4);
  float* scanC = (float*)alloc((size_t)Bb * 768 * CH * 4);

  float* outx = (float*)d_out;
  float* outy = outx + (size_t)TX * 768;

  hipMemsetAsync(accums, 0, 16 * sizeof(float), stream);
  for (int i = 0; i < 16; ++i) {
    const size_t n = (size_t)qw[i].K * qw[i].O;
    absmean_partial_kernel<<<512, 256, 0, stream>>>(qw[i].w, n, accums + i);
    ternarize_pack_kernel<<<2048, 256, 0, stream>>>(qw[i].w, qw[i].K, qw[i].O,
                                                    accums + i, wscales + i,
                                                    qw[i].wt);
  }

  silu_kernel<<<(Bb * 768 + 255) / 256, 256, 0, stream>>>(c_in, cs,
                                                          (size_t)Bb * 768);
  matvec_kernel<<<dim3(3, Bb), 256, 0, stream>>>(cs, ada_w_in, nullptr, h1, 768, 768);
  matvec_kernel<<<dim3(16, Bb), 256, 0, stream>>>(h1, ada_wgate, nullptr, y2, 768, 4096);
  glu_ada_kernel<<<(Bb * 2048 + 255) / 256, 256, 0, stream>>>(y2, h2g, (size_t)Bb * 2048);
  matvec_kernel<<<dim3(3, Bb), 256, 0, stream>>>(h2g, ada_wdown, nullptr, h2, 2048, 768);
  matvec_kernel<<<dim3(45, Bb), 256, 0, stream>>>(h2, ada_woutp, ada_boutp, h3, 768, 11520);
  rmsnormw_kernel<<<Bb, 256, 0, stream>>>(h3, ada_normw, h4, 11520);
  matvec_kernel<<<dim3(45, Bb), 256, 0, stream>>>(h4, ada_wout2, ada_bout2, mods, 11520, 11520);

  hipMemcpyAsync(outx, x_in, (size_t)TX * 768 * 4, hipMemcpyDeviceToDevice, stream);
  hipMemcpyAsync(outy, y_in, (size_t)TY * 768 * 4, hipMemcpyDeviceToDevice, stream);

  auto gemm = [&](const int8_t* wt, const float* sc, float* out, int T, int K,
                  int O, int mode) {
    dim3 g((T + 31) / 32, (O + 255) / 256);
    if (mode)
      bitlin_gemm_iu8<1><<<g, 256, 0, stream>>>(xq, asc, wt, sc, out, T, K, O);
    else
      bitlin_gemm_iu8<0><<<g, 256, 0, stream>>>(xq, asc, wt, sc, out, T, K, O);
  };
  auto bitlin = [&](const float* X, int T, int K, int wi, float* out, int O) {
    rms_quant_kernel<<<T, 256, 0, stream>>>(X, K, xq, asc, 1e-8f);
    gemm(qw[wi].wt, wscales + wi, out, T, K, O, 0);
  };
  auto scan = [&](const float* F, float* IH, int S) {
    const int tot = Bb * 768 * CH;
    hgrn_chunk_scan_kernel<<<(tot + 255) / 256, 256, 0, stream>>>(F, IH, scanP,
                                                                  Bb, S, CH);
    hgrn_carry_kernel<<<(Bb * 768 + 255) / 256, 256, 0, stream>>>(IH, scanP,
                                                                  scanC, Bb, S,
                                                                  CH);
    hgrn_apply_kernel<<<(tot + 255) / 256, 256, 0, stream>>>(F, IH, scanC, Bb,
                                                             S, CH);
  };

  auto hgrn = [&](float* X, int S, int T, int w0, const float* gnorm,
                  int chunk0) {
    ln_modulate_kernel<<<T, 256, 0, stream>>>(X, mods, chunk0, chunk0 + 1, bufXM, S);
    rms_quant_kernel<<<T, 256, 0, stream>>>(bufXM, 768, xq, asc, 1e-8f);
    gemm(qw[w0 + 0].wt, wscales + w0 + 0, bufI, T, 768, 768, 0);
    gemm(qw[w0 + 1].wt, wscales + w0 + 1, bufF, T, 768, 768, 0);
    gemm(qw[w0 + 2].wt, wscales + w0 + 2, bufG, T, 768, 768, 0);
    const size_t n = (size_t)T * 768;
    sigmoid_inplace_kernel<<<(unsigned)((n + 255) / 256), 256, 0, stream>>>(bufF, n);
    gate_combine_kernel<<<(unsigned)((n + 255) / 256), 256, 0, stream>>>(bufI, bufF, n);
    scan(bufF, bufI, S);
    rms_gate_kernel<<<T, 256, 0, stream>>>(bufI, bufG, gnorm);
    bitlin(bufG, T, 768, w0 + 3, bufF, 768);
    residual_gate_kernel<<<(unsigned)((n + 255) / 256), 256, 0, stream>>>(
        X, bufF, mods, chunk0 + 2, S, n);
  };

  auto mlp = [&](float* X, int S, int T, int wg, int wd, int chunk0) {
    ln_modulate_kernel<<<T, 256, 0, stream>>>(X, mods, chunk0, chunk0 + 1, bufXM, S);
    rms_quant_kernel<<<T, 256, 0, stream>>>(bufXM, 768, xq, asc, 1e-8f);
    gemm(qw[wg].wt, wscales + wg, bufBig, T, 768, 3072, 0);
    gemm(qw[wg].wt + (size_t)3072 * 768, wscales + wg, bufBig, T, 768, 3072, 1);
    bitlin(bufBig, T, 3072, wd, bufI, 768);
    const size_t n = (size_t)T * 768;
    residual_gate_kernel<<<(unsigned)((n + 255) / 256), 256, 0, stream>>>(
        X, bufI, mods, chunk0 + 2, S, n);
  };

  hgrn(outx, Nn, TX,   0, gnorm_x,   0);

  ln_modulate_kernel<<<TX, 256, 0, stream>>>(outx, mods, 3, 4, bufXM, Nn);
  bitlin(bufXM, TX, 768,   8, bufI, 768);
  rms_quant_kernel<<<TY, 256, 0, stream>>>(outy, 768, xq, asc, 1e-8f);
  gemm(qw[9].wt, wscales + 9, bufK, TY, 768, 768, 0);
  gemm(qw[10].wt, wscales + 10, bufV, TY, 768, 768, 0);
  to_f16_heads_kernel<<<(unsigned)(((size_t)TX * 768 + 255) / 256), 256, 0,
                        stream>>>(bufI, qh, Nn, (size_t)TX * 768);
  to_f16_heads_kernel<<<(unsigned)(((size_t)TY * 768 + 255) / 256), 256, 0,
                        stream>>>(bufK, kh, Mm, (size_t)TY * 768);
  to_f16_headsT_kernel<<<(unsigned)(((size_t)TY * 768 + 255) / 256), 256, 0,
                         stream>>>(bufV, vh, Mm, (size_t)TY * 768);
  cross_attn_kernel<<<(Bb * 12 * (Nn / 16)) / 8, 256, 0, stream>>>(
      qh, kh, vh, bufG, Bb, Nn, Mm);
  bitlin(bufG, TX, 768,   11, bufF, 768);
  residual_gate_kernel<<<(unsigned)(((size_t)TX * 768 + 255) / 256), 256, 0,
                         stream>>>(outx, bufF, mods, 5, Nn, (size_t)TX * 768);

  mlp(outx, Nn, TX,   12,   13,   6);

  hgrn(outy, Mm, TY,   4, gnorm_y,   9);
  mlp(outy, Mm, TY,   14,   15,   12);
}
